// Model_72421738545362
// MI455X (gfx1250) — matrix operand order repaired and checked by reading, not compiled, not run on hardware
//
#include <hip/hip_runtime.h>

#define HD 192
#define NB 32
#define WW 50
#define VV 32000
#define ZD 768

typedef __attribute__((ext_vector_type(16))) __bf16 v16bf;
typedef __attribute__((ext_vector_type(8)))  float  v8f;

// ---------------------------------------------------------------------------
// WMMA fragment helpers (contiguous k order used by this code for A and B, wave32)
// A (16x32, MxK): lanes 0-15 row M=lane, K-base 0; lanes 16-31 row M=lane-16,
//   K-base 16. halves 0..15 -> K=kb..kb+15 (matches B).
//   A is staged in LDS as fp32 (filled by async copies), converted to bf16
//   at fragment-load time.
// B (32x16, KxN): lanes 0-15 col N=lane, K=0..15; lanes 16-31 K=16..31,
//   sequential halves. Weights stored transposed (N-major) so this is one
//   contiguous 32B load per lane.
// ---------------------------------------------------------------------------
__device__ __forceinline__ v16bf load_a_frag(const float* A, int K, int mtile,
                                             int kt, int lane) {
  int m  = mtile * 16 + (lane & 15);
  int kb = kt * 32 + ((lane >> 4) << 4);
  const float* p0 = A + m * K + kb;
  const float* p1 = p0 + 8;
  v16bf a;
#pragma unroll
  for (int i = 0; i < 8; ++i) { a[i] = (__bf16)p0[i]; a[i + 8] = (__bf16)p1[i]; }
  return a;
}

__device__ __forceinline__ v16bf load_b_frag(const __bf16* WT, int K, int nbase,
                                             int kt, int lane) {
  int n  = nbase + (lane & 15);
  int kb = kt * 32 + ((lane >> 4) << 4);
  return *(const v16bf*)(WT + (size_t)n * K + kb);
}

// ---------------------------------------------------------------------------
// Z(32 x N) (+)= [src0 | src1](32 x (K0+K1)) @ WT^T  [+ bias]
// WT is bf16, transposed N-major: WT[n*K + k]. A tile staged fp32 in LDS via
// CDNA5 async global->LDS copies (ASYNCcnt), then consumed by bf16 WMMA.
// 256 threads = 8 waves; each wave owns one 16x16 output tile
// (2 M-tiles x 4 N-tiles per block, 64 columns per block). K0, K1 mult. of 4,
// K0+K1 <= 384, all row bases 16B aligned.
// ---------------------------------------------------------------------------
__global__ void wmma_gemm_cat(const float* __restrict__ src0, int ld0, int K0,
                              const float* __restrict__ src1, int ld1, int K1,
                              const __bf16* __restrict__ WT,
                              const float* __restrict__ bias,
                              float* __restrict__ Z, int N, int accumulate) {
  __shared__ float Alds[32 * 384];
  const int K = K0 + K1;
  const int tid = threadIdx.x;

  // ---- async fill: 16B chunks, global -> LDS without VGPR round-trip ----
  {
    unsigned ldsbase = (unsigned)(uintptr_t)(void*)Alds;
    int rowchunks = K >> 2;            // 16B chunks per A row
    int nchunks   = 32 * rowchunks;
    for (int c = tid; c < nchunks; c += 256) {
      int r  = c / rowchunks;
      int k4 = (c - r * rowchunks) << 2;  // float offset within row
      const float* gsrc = (k4 < K0) ? (src0 + (size_t)r * ld0 + k4)
                                    : (src1 + (size_t)r * ld1 + (k4 - K0));
      unsigned lds_addr = ldsbase + ((unsigned)(r * K + k4) << 2);
      unsigned long long ga = (unsigned long long)(uintptr_t)gsrc;
      asm volatile("global_load_async_to_lds_b128 %0, %1, off"
                   :: "v"(lds_addr), "v"(ga) : "memory");
    }
    asm volatile("s_wait_asynccnt 0x0" ::: "memory");
  }
  __syncthreads();

  const int wave  = tid >> 5, lane = tid & 31;
  const int mtile = wave & 1;
  const int nbase = (blockIdx.x * 4 + (wave >> 1)) << 4;

  v8f acc = {};
  const int nkt = K >> 5;
  for (int kt = 0; kt < nkt; ++kt) {
    v16bf a = load_a_frag(Alds, K, mtile, kt, lane);
    v16bf b = load_b_frag(WT, K, nbase, kt, lane);
    acc = __builtin_amdgcn_wmma_f32_16x16x32_bf16(false, a, false, b,
                                                  (short)0, acc, false, false);
  }

  const int nn   = nbase + (lane & 15);
  const int mrow = mtile * 16 + ((lane >> 4) << 3);
  if (accumulate) {
#pragma unroll
    for (int v = 0; v < 8; ++v)
      Z[(size_t)(mrow + v) * N + nn] += acc[v];
  } else {
    float bv = bias ? bias[nn] : 0.0f;
#pragma unroll
    for (int v = 0; v < 8; ++v)
      Z[(size_t)(mrow + v) * N + nn] = acc[v] + bv;
  }
}

// fp32 [A(K0 x N); B(K1 x N)] -> bf16 transposed (N-major):
// out[n*(K0+K1)+k] = (k<K0) ? A[k*N+n] : B[(k-K0)*N+n]
__global__ void transpose_concat_bf16(const float* __restrict__ A, int K0,
                                      const float* __restrict__ B, int K1,
                                      int N, __bf16* __restrict__ out) {
  size_t idx = (size_t)blockIdx.x * 256 + threadIdx.x;
  int K = K0 + K1;
  if (idx >= (size_t)N * K) return;
  int n = (int)(idx / K);
  int k = (int)(idx - (size_t)n * K);
  float v = (k < K0) ? A[(size_t)k * N + n] : B[(size_t)(k - K0) * N + n];
  out[idx] = (__bf16)v;
}

__global__ void zero_f32(float* p, int n) {
  int i = blockIdx.x * 256 + threadIdx.x;
  if (i < n) p[i] = 0.0f;
}

__global__ void gather_embed(const int* __restrict__ tok,
                             const float* __restrict__ emb,
                             float* __restrict__ dst, int total) {
  int idx = blockIdx.x * 256 + threadIdx.x;
  if (idx >= total) return;
  int i = idx / HD, j = idx - i * HD;
  dst[idx] = emb[(size_t)tok[i] * HD + j];
}

__device__ __forceinline__ float sigmoidf(float x) {
  return 1.0f / (1.0f + expf(-x));
}

__global__ void lstm_gates(const float* __restrict__ Z, float* __restrict__ h,
                           float* __restrict__ c, float* __restrict__ out_h) {
  int idx = blockIdx.x * 256 + threadIdx.x;
  if (idx >= NB * HD) return;
  int b = idx / HD, j = idx - b * HD;
  const float* z = Z + (size_t)b * ZD;
  float ig = sigmoidf(z[j]);
  float fg = sigmoidf(z[HD + j]);
  float gg = tanhf(z[2 * HD + j]);
  float og = sigmoidf(z[3 * HD + j]);
  float cn = fg * c[idx] + ig * gg;
  float hn = og * tanhf(cn);
  c[idx] = cn;
  h[idx] = hn;
  if (out_h) out_h[idx] = hn;
}

// enc_out[b][w][:] = outs_f[w][b][:] + outs_b[49-w][b][:]
__global__ void combine_encout(const float* __restrict__ of,
                               const float* __restrict__ ob,
                               float* __restrict__ enc_out) {
  int idx = blockIdx.x * 256 + threadIdx.x;
  if (idx >= NB * WW * HD) return;
  int b = idx / (WW * HD);
  int rem = idx - b * (WW * HD);
  int w = rem / HD, j = rem - w * HD;
  enc_out[idx] = of[((size_t)w * NB + b) * HD + j] +
                 ob[((size_t)(WW - 1 - w) * NB + b) * HD + j];
}

__global__ void add_states(const float* a, const float* b2, float* o, int n) {
  int i = blockIdx.x * 256 + threadIdx.x;
  if (i < n) o[i] = a[i] + b2[i];
}

// One block per batch row: q = h@Wq, attention scores + softmax over W+1,
// context c_t, writes betas/g and the decoder-LSTM input xcat=[demb_t, c_t].
__global__ void dec_attn(const float* __restrict__ hd,
                         const float* __restrict__ Wq,
                         const float* __restrict__ enc_out,
                         const float* __restrict__ sentinel,
                         const float* __restrict__ a_bias,
                         const float* __restrict__ dec_emb,
                         float* __restrict__ betas, float* __restrict__ gptr,
                         float* __restrict__ ct, float* __restrict__ xcat,
                         int t) {
  int b = blockIdx.x;
  __shared__ float qs[HD];
  __shared__ float sc[WW + 1];
  int tid = threadIdx.x;
  const float* hrow = hd + (size_t)b * HD;

  for (int j = tid; j < HD; j += 256) {
    float acc = 0.0f;
    for (int k = 0; k < HD; ++k) acc += hrow[k] * Wq[k * HD + j];
    qs[j] = acc;
  }
  __syncthreads();

  int wave = tid >> 5, lane = tid & 31;
  for (int w = wave; w < WW + 1; w += 8) {
    const float* frow =
        (w < WW) ? (enc_out + ((size_t)b * WW + w) * HD) : sentinel;
    float p = 0.0f;
    for (int j = lane; j < HD; j += 32) p += tanhf(frow[j] * qs[j]);
    for (int off = 16; off > 0; off >>= 1) p += __shfl_xor(p, off, 32);
    if (lane == 0) sc[w] = p + a_bias[t * (WW + 1) + w];
  }
  __syncthreads();

  if (tid == 0) {
    float m = sc[0];
    for (int w = 1; w < WW + 1; ++w) m = fmaxf(m, sc[w]);
    float s = 0.0f;
    for (int w = 0; w < WW + 1; ++w) { sc[w] = expf(sc[w] - m); s += sc[w]; }
    float inv = 1.0f / s;
    for (int w = 0; w < WW + 1; ++w) sc[w] *= inv;
    gptr[b] = sc[WW];
  }
  __syncthreads();

  if (tid < WW) betas[b * WW + tid] = sc[tid];

  for (int j = tid; j < HD; j += 256) {
    float acc = 0.0f;
    for (int w = 0; w < WW; ++w)
      acc += sc[w] * enc_out[((size_t)b * WW + w) * HD + j];
    ct[(size_t)b * HD + j] = acc;
    xcat[(size_t)b * 2 * HD + j] = dec_emb[((size_t)b * WW + t) * HD + j];
    xcat[(size_t)b * 2 * HD + HD + j] = acc;
  }
}

__global__ void row_softmax_stats(const float* __restrict__ L, int N,
                                  float* __restrict__ mx,
                                  float* __restrict__ sm) {
  int b = blockIdx.x;
  __shared__ float red[256];
  const float* row = L + (size_t)b * N;
  float m = -3.402823466e+38f;
  for (int i = threadIdx.x; i < N; i += 256) m = fmaxf(m, row[i]);
  red[threadIdx.x] = m;
  __syncthreads();
  for (int s = 128; s > 0; s >>= 1) {
    if (threadIdx.x < s)
      red[threadIdx.x] = fmaxf(red[threadIdx.x], red[threadIdx.x + s]);
    __syncthreads();
  }
  float rowm = red[0];
  __syncthreads();
  float acc = 0.0f;
  for (int i = threadIdx.x; i < N; i += 256) acc += expf(row[i] - rowm);
  red[threadIdx.x] = acc;
  __syncthreads();
  for (int s = 128; s > 0; s >>= 1) {
    if (threadIdx.x < s) red[threadIdx.x] += red[threadIdx.x + s];
    __syncthreads();
  }
  if (threadIdx.x == 0) { mx[b] = rowm; sm[b] = red[0]; }
}

__global__ void finalize_probs(const float* __restrict__ L,
                               const float* __restrict__ mx,
                               const float* __restrict__ sm,
                               const float* __restrict__ g,
                               float* __restrict__ out, int t) {
  int idx = blockIdx.x * 256 + threadIdx.x;
  if (idx >= NB * VV) return;
  int b = idx / VV, v = idx - b * VV;
  float p = expf(L[idx] - mx[b]) / sm[b];
  out[((size_t)b * (WW - 1) + t) * VV + v] = g[b] * p;
}

__global__ void ptr_scatter(const int* __restrict__ enc_in,
                            const float* __restrict__ betas,
                            const float* __restrict__ g,
                            float* __restrict__ out, int t) {
  int idx = blockIdx.x * 256 + threadIdx.x;
  if (idx >= NB * WW) return;
  int b = idx / WW;
  int tok = enc_in[idx];
  atomicAdd(out + ((size_t)b * (WW - 1) + t) * VV + tok,
            (1.0f - g[b]) * betas[idx]);
}

// ---------------------------------------------------------------------------
extern "C" void kernel_launch(void* const* d_in, const int* in_sizes, int n_in,
                              void* d_out, int out_size, void* d_ws,
                              size_t ws_size, hipStream_t stream) {
  (void)in_sizes; (void)n_in; (void)out_size; (void)ws_size;
  const int*   enc_in   = (const int*)d_in[0];
  const int*   dec_in   = (const int*)d_in[1];
  const float* emb      = (const float*)d_in[2];
  const float* Wf       = (const float*)d_in[3];
  const float* Uf       = (const float*)d_in[4];
  const float* bf       = (const float*)d_in[5];
  const float* Wb       = (const float*)d_in[6];
  const float* Ub       = (const float*)d_in[7];
  const float* bb       = (const float*)d_in[8];
  const float* Wd       = (const float*)d_in[9];
  const float* Ud       = (const float*)d_in[10];
  const float* bd       = (const float*)d_in[11];
  const float* Wq       = (const float*)d_in[12];
  const float* Wdense   = (const float*)d_in[13];
  const float* bdense   = (const float*)d_in[14];
  const float* sentinel = (const float*)d_in[15];
  const float* a_bias   = (const float*)d_in[16];
  float* out = (float*)d_out;

  char* wsp = (char*)d_ws;
  auto carve = [&](size_t bytes) -> char* {
    char* p = wsp;
    wsp += (bytes + 255) & ~(size_t)255;
    return p;
  };

  // bf16 transposed (N-major) weights; LSTM input+recurrent pre-concatenated
  __bf16* WUfT = (__bf16*)carve((size_t)ZD * 384 * 2);  // [Wf;Uf]^T 768x384
  __bf16* WUbT = (__bf16*)carve((size_t)ZD * 384 * 2);  // [Wb;Ub]^T 768x384
  __bf16* WdT  = (__bf16*)carve((size_t)ZD * 384 * 2);  // Wd^T     768x384
  __bf16* UdT  = (__bf16*)carve((size_t)ZD * HD * 2);   // Ud^T     768x192
  __bf16* WdnT = (__bf16*)carve((size_t)VV * 384 * 2);  // Wdense^T 32000x384
  // fp32 activations / state
  float* enc_emb = (float*)carve((size_t)NB * WW * HD * 4);
  float* dec_emb = (float*)carve((size_t)NB * WW * HD * 4);
  float* outs_f  = (float*)carve((size_t)WW * NB * HD * 4);
  float* outs_b  = (float*)carve((size_t)WW * NB * HD * 4);
  float* enc_out = (float*)carve((size_t)NB * WW * HD * 4);
  float* hf = (float*)carve((size_t)NB * HD * 4);
  float* cf = (float*)carve((size_t)NB * HD * 4);
  float* hb = (float*)carve((size_t)NB * HD * 4);
  float* cb = (float*)carve((size_t)NB * HD * 4);
  float* hd = (float*)carve((size_t)NB * HD * 4);
  float* cd = (float*)carve((size_t)NB * HD * 4);
  float* Zf = (float*)carve((size_t)NB * ZD * 4);
  float* Zb = (float*)carve((size_t)NB * ZD * 4);
  float* betas  = (float*)carve((size_t)NB * WW * 4);
  float* gv     = (float*)carve((size_t)NB * 4);
  float* ctb    = (float*)carve((size_t)NB * HD * 4);
  float* xcat   = (float*)carve((size_t)NB * 2 * HD * 4);
  float* logits = (float*)carve((size_t)NB * VV * 4);
  float* rowmx  = (float*)carve((size_t)NB * 4);
  float* rowsm  = (float*)carve((size_t)NB * 4);

  auto gridN = [](size_t n) { return (unsigned)((n + 255) / 256); };

  // --- weight prep (fp32 -> bf16 transposed, LSTM K-concat folded in) ---
  transpose_concat_bf16<<<gridN((size_t)ZD * 384), 256, 0, stream>>>(
      Wf, HD, Uf, HD, ZD, WUfT);
  transpose_concat_bf16<<<gridN((size_t)ZD * 384), 256, 0, stream>>>(
      Wb, HD, Ub, HD, ZD, WUbT);
  transpose_concat_bf16<<<gridN((size_t)ZD * 384), 256, 0, stream>>>(
      Wd, 2 * HD, nullptr, 0, ZD, WdT);
  transpose_concat_bf16<<<gridN((size_t)ZD * HD), 256, 0, stream>>>(
      Ud, HD, nullptr, 0, ZD, UdT);
  transpose_concat_bf16<<<gridN((size_t)VV * 384), 256, 0, stream>>>(
      Wdense, 2 * HD, nullptr, 0, VV, WdnT);

  // --- embeddings ---
  gather_embed<<<gridN(NB * WW * HD), 256, 0, stream>>>(enc_in, emb, enc_emb,
                                                        NB * WW * HD);
  gather_embed<<<gridN(NB * WW * HD), 256, 0, stream>>>(dec_in, emb, dec_emb,
                                                        NB * WW * HD);

  // --- zero init states ---
  zero_f32<<<gridN(NB * HD), 256, 0, stream>>>(hf, NB * HD);
  zero_f32<<<gridN(NB * HD), 256, 0, stream>>>(cf, NB * HD);
  zero_f32<<<gridN(NB * HD), 256, 0, stream>>>(hb, NB * HD);
  zero_f32<<<gridN(NB * HD), 256, 0, stream>>>(cb, NB * HD);

  const unsigned gZ = ZD / 64;  // 12 blocks for 768-wide GEMMs
  const unsigned gV = VV / 64;  // 500 blocks for 32000-wide GEMMs

  // --- bidirectional encoder scan ---
  for (int s = 0; s < WW; ++s) {
    // forward: z = [x_s, hf] @ [Wf;Uf] + bf
    wmma_gemm_cat<<<gZ, 256, 0, stream>>>(enc_emb + (size_t)s * HD, WW * HD, HD,
                                          hf, HD, HD, WUfT, bf, Zf, ZD, 0);
    lstm_gates<<<gridN(NB * HD), 256, 0, stream>>>(
        Zf, hf, cf, outs_f + (size_t)s * NB * HD);
    // backward: x = enc_emb[:, W-1-s]
    wmma_gemm_cat<<<gZ, 256, 0, stream>>>(enc_emb + (size_t)(WW - 1 - s) * HD,
                                          WW * HD, HD, hb, HD, HD, WUbT, bb, Zb,
                                          ZD, 0);
    lstm_gates<<<gridN(NB * HD), 256, 0, stream>>>(
        Zb, hb, cb, outs_b + (size_t)s * NB * HD);
  }

  combine_encout<<<gridN((size_t)NB * WW * HD), 256, 0, stream>>>(outs_f,
                                                                  outs_b,
                                                                  enc_out);
  add_states<<<gridN(NB * HD), 256, 0, stream>>>(hf, hb, hd, NB * HD);
  add_states<<<gridN(NB * HD), 256, 0, stream>>>(cf, cb, cd, NB * HD);

  // --- decoder scan ---
  for (int t = 0; t < WW - 1; ++t) {
    dec_attn<<<NB, 256, 0, stream>>>(hd, Wq, enc_out, sentinel, a_bias, dec_emb,
                                     betas, gv, ctb, xcat, t);
    // LSTM: z = [demb,c_t]@Wd + h@Ud + bd   (K=576 split 384+192)
    wmma_gemm_cat<<<gZ, 256, 0, stream>>>(xcat, 2 * HD, 2 * HD, nullptr, 0, 0,
                                          WdT, bd, Zf, ZD, 0);
    wmma_gemm_cat<<<gZ, 256, 0, stream>>>(hd, HD, HD, nullptr, 0, 0, UdT,
                                          nullptr, Zf, ZD, 1);
    lstm_gates<<<gridN(NB * HD), 256, 0, stream>>>(Zf, hd, cd, nullptr);
    // logits = [h, c_t] @ Wdense + bdense  (single K=384 pass)
    wmma_gemm_cat<<<gV, 256, 0, stream>>>(hd, HD, HD, ctb, HD, HD, WdnT, bdense,
                                          logits, VV, 0);
    row_softmax_stats<<<NB, 256, 0, stream>>>(logits, VV, rowmx, rowsm);
    finalize_probs<<<gridN((size_t)NB * VV), 256, 0, stream>>>(
        logits, rowmx, rowsm, gv, out, t);
    ptr_scatter<<<gridN(NB * WW), 256, 0, stream>>>(enc_in, betas, gv, out, t);
  }
}
